// Attention_18837726560765
// MI455X (gfx1250) — matrix operand order repaired and checked by reading, not compiled, not run on hardware
//
#include <hip/hip_runtime.h>

// Fused attention pipeline for MI455X (gfx1250), wave32, BF16 WMMA.
//   Q = q@Wq+bq ; K = x@Wk+bk ; V = x@Wv+bv   (bf16 out, f32 accum)
//   S = sigmoid((Q@K^T)/32)                    (bf16, 32MB, L2-resident)
//   out = S@V                                  (f32)
// All GEMMs are TN (B stored [N,K]) so LDS staging is pure ds_store_b128 /
// global_load_async_to_lds_b128.  Workspace use: ~86 MB.

typedef __bf16 bf16_t;
typedef __attribute__((ext_vector_type(4)))  __bf16 v4bf;
typedef __attribute__((ext_vector_type(8)))  __bf16 v8bf;
typedef __attribute__((ext_vector_type(16))) __bf16 v16bf;
typedef __attribute__((ext_vector_type(8)))  float  v8f;
typedef __attribute__((__vector_size__(16))) int    v4i_t;

#define LQ 4096
#define LKS 4096
#define DIN 1024
#define DOUT 1024

#define BM 128
#define BN 128
#define BKT 64
#define KST 72   // LDS row stride (bf16): 144B row = 36 dwords -> conflict-free, 16B aligned

#if defined(__has_builtin)
#if __has_builtin(__builtin_amdgcn_global_load_async_to_lds_b128)
#define USE_ASYNC 1
#endif
#endif
#ifndef USE_ASYNC
#define USE_ASYNC 0
#endif

// ---------------------------------------------------------------- helpers
__device__ inline v8f wmma_bf16(v16bf a, v16bf b, v8f c) {
  return __builtin_amdgcn_wmma_f32_16x16x32_bf16(false, a, false, b, (short)0, c,
                                                 false, false);
}

// A fragment (16x32, 16-bit): lane holds row lane&15; elems 0..7 = K kb..kb+7,
// elems 8..15 = K 16+kb..  (kb = (lane>>4)*8)
__device__ inline v16bf frag_a(const bf16_t* p) {
  v8bf lo = *(const v8bf*)p;
  v8bf hi = *(const v8bf*)(p + 16);
  v16bf r;
#pragma unroll
  for (int e = 0; e < 8; ++e) { r[e] = lo[e]; r[e + 8] = hi[e]; }
  return r;
}
// B fragment (32x16): lane holds col lane&15; elems 0..7 = K kb..kb+7,
// elems 8..15 = K 16+kb..  (kb = (lane>>4)*8)
__device__ inline v16bf frag_b(const bf16_t* p) {
  v8bf lo = *(const v8bf*)p;
  v8bf hi = *(const v8bf*)(p + 16);
  v16bf r;
#pragma unroll
  for (int e = 0; e < 8; ++e) { r[e] = lo[e]; r[e + 8] = hi[e]; }
  return r;
}

#if USE_ASYNC
__device__ inline void async_cp16(const bf16_t* g, bf16_t* l) {
  __builtin_amdgcn_global_load_async_to_lds_b128(
      (__attribute__((address_space(1))) v4i_t*)(bf16_t*)g,
      (__attribute__((address_space(3))) v4i_t*)l,
      0, 0);
}
__device__ inline void wait_async0() {
#if __has_builtin(__builtin_amdgcn_s_wait_asynccnt)
  __builtin_amdgcn_s_wait_asynccnt(0);
#else
  asm volatile("s_wait_asynccnt 0x0" ::: "memory");
#endif
}
#endif

// ---------------------------------------------------------------- converts
__global__ __launch_bounds__(256) void cvt_f32_to_bf16(
    const float* __restrict__ s, bf16_t* __restrict__ d, int n) {
  int i = (blockIdx.x * 256 + threadIdx.x) * 4;
  if (i >= n) return;
  float4 v = *(const float4*)(s + i);
  v4bf o;
  o[0] = (bf16_t)v.x; o[1] = (bf16_t)v.y; o[2] = (bf16_t)v.z; o[3] = (bf16_t)v.w;
  *(v4bf*)(d + i) = o;
}

// out[c][r] = (bf16) in[r][c];  in: f32 [R,C]; 32x32 LDS tiles, coalesced both sides
__global__ __launch_bounds__(256) void cvt_tr_f32_bf16(
    const float* __restrict__ in, bf16_t* __restrict__ out, int R, int C) {
  __shared__ bf16_t t[32 * 36];
  const int bi = blockIdx.y * 32;           // row base (R)
  const int bj = blockIdx.x * 32;           // col base (C)
  const int tr = threadIdx.x >> 3;          // 0..31
  const int tc = (threadIdx.x & 7) * 4;     // 0,4,..,28
  float4 v = *(const float4*)(in + (size_t)(bi + tr) * C + bj + tc);
  t[tr * 36 + tc + 0] = (bf16_t)v.x;
  t[tr * 36 + tc + 1] = (bf16_t)v.y;
  t[tr * 36 + tc + 2] = (bf16_t)v.z;
  t[tr * 36 + tc + 3] = (bf16_t)v.w;
  __syncthreads();
  v4bf o;
#pragma unroll
  for (int j = 0; j < 4; ++j) o[j] = t[(tc + j) * 36 + tr];
  *(v4bf*)(out + (size_t)(bj + tr) * R + bi + tc) = o;
}

// out[c][r] = in[r][c];  bf16 [R,C] -> bf16 [C,R]
__global__ __launch_bounds__(256) void tr_bf16(
    const bf16_t* __restrict__ in, bf16_t* __restrict__ out, int R, int C) {
  __shared__ bf16_t t[32 * 36];
  const int bi = blockIdx.y * 32;
  const int bj = blockIdx.x * 32;
  const int tr = threadIdx.x >> 3;
  const int tc = (threadIdx.x & 7) * 4;
  v4bf v = *(const v4bf*)(in + (size_t)(bi + tr) * C + bj + tc);
#pragma unroll
  for (int j = 0; j < 4; ++j) t[tr * 36 + tc + j] = v[j];
  __syncthreads();
  v4bf o;
#pragma unroll
  for (int j = 0; j < 4; ++j) o[j] = t[(tc + j) * 36 + tr];
  *(v4bf*)(out + (size_t)(bj + tr) * R + bi + tc) = o;
}

// ---------------------------------------------------------------- TN GEMM
// C[M,N] = A[M,K] @ B[N,K]^T + epilogue
//   MODE 0: +bias[n] -> bf16.  MODE 1: sigmoid(acc*scale) -> bf16.  MODE 2: f32.
// Tile 128x128x64; 8 waves (4x2); each wave 32x64 (2x4 accumulators);
// 16 v_wmma per k-step; async double-buffered (or reg-prefetch) staging.
template <int MODE>
__global__ __launch_bounds__(256) void gemm_tn(
    const bf16_t* __restrict__ A, const bf16_t* __restrict__ B,
    const float* __restrict__ bias, void* __restrict__ C,
    int M, int N, int Kdim, float scale) {
#if USE_ASYNC
  __shared__ bf16_t As[2][BM * KST];
  __shared__ bf16_t Bs[2][BN * KST];
#else
  __shared__ bf16_t As[1][BM * KST];
  __shared__ bf16_t Bs[1][BN * KST];
#endif
  const int tid   = threadIdx.x;
  const int lane  = tid & 31;
  const int wave  = tid >> 5;
  const int wm    = wave & 3;          // 4 row-groups of 32
  const int wn    = wave >> 2;         // 2 col-groups of 64
  const int lhalf = lane >> 4;
  const int lrow  = lane & 15;
  const int bm0 = blockIdx.y * BM;
  const int bn0 = blockIdx.x * BN;

  // staging: tile = 128 rows x 64 k bf16 = 1024 chunks of 16B; 4 chunks/thread
  const int r0  = tid >> 3;            // rows r0, r0+32, r0+64, r0+96
  const int kc0 = (tid & 7) * 8;

  v8f acc[2][4] = {};

  auto compute_tile = [&](int buf) {
#pragma unroll
    for (int s = 0; s < 2; ++s) {      // two K=32 sub-steps
      v16bf af[2], bfv[4];
#pragma unroll
      for (int mi = 0; mi < 2; ++mi)
        af[mi] = frag_a(&As[buf][(wm * 32 + mi * 16 + lrow) * KST + s * 32 + lhalf * 8]);
#pragma unroll
      for (int ni = 0; ni < 4; ++ni)
        bfv[ni] = frag_b(&Bs[buf][(wn * 64 + ni * 16 + lrow) * KST + s * 32 + lhalf * 8]);
#pragma unroll
      for (int mi = 0; mi < 2; ++mi)
#pragma unroll
        for (int ni = 0; ni < 4; ++ni)
          acc[mi][ni] = wmma_bf16(af[mi], bfv[ni], acc[mi][ni]);
    }
  };

  const int nsteps = Kdim / BKT;

#if USE_ASYNC
  auto issue_tile = [&](int k0, int buf) {
#pragma unroll
    for (int it = 0; it < 4; ++it) {
      int r = r0 + it * 32;
      async_cp16(A + (size_t)(bm0 + r) * Kdim + k0 + kc0, &As[buf][r * KST + kc0]);
      async_cp16(B + (size_t)(bn0 + r) * Kdim + k0 + kc0, &Bs[buf][r * KST + kc0]);
    }
  };
  issue_tile(0, 0);
  for (int i = 0; i < nsteps; ++i) {
    wait_async0();                       // this wave's batch i landed in LDS
    __syncthreads();                     // everyone's batch i landed
    if (i + 1 < nsteps) issue_tile((i + 1) * BKT, (i + 1) & 1);  // overlaps compute
    compute_tile(i & 1);
    __syncthreads();                     // buf reads done before it is re-filled
  }
#else
  uint4 ra[4], rb[4];
  auto load_regs = [&](int k0) {
#pragma unroll
    for (int it = 0; it < 4; ++it) {
      int r = r0 + it * 32;
      ra[it] = *(const uint4*)(A + (size_t)(bm0 + r) * Kdim + k0 + kc0);
      rb[it] = *(const uint4*)(B + (size_t)(bn0 + r) * Kdim + k0 + kc0);
    }
  };
  load_regs(0);
  for (int i = 0; i < nsteps; ++i) {
#pragma unroll
    for (int it = 0; it < 4; ++it) {
      int r = r0 + it * 32;
      *(uint4*)(&As[0][r * KST + kc0]) = ra[it];
      *(uint4*)(&Bs[0][r * KST + kc0]) = rb[it];
    }
    __syncthreads();
    if (i + 1 < nsteps) load_regs((i + 1) * BKT);   // overlaps compute
    compute_tile(0);
    __syncthreads();
  }
#endif

  // epilogue: C/D layout elem r -> M = (lane>>4)*8 + r, N = lane&15
#pragma unroll
  for (int mi = 0; mi < 2; ++mi)
#pragma unroll
    for (int ni = 0; ni < 4; ++ni)
#pragma unroll
      for (int r = 0; r < 8; ++r) {
        int gm = bm0 + wm * 32 + mi * 16 + lhalf * 8 + r;
        int gn = bn0 + wn * 64 + ni * 16 + lrow;
        float v = acc[mi][ni][r];
        if (MODE == 0) {
          v += bias[gn];
          ((bf16_t*)C)[(size_t)gm * N + gn] = (bf16_t)v;
        } else if (MODE == 1) {
          v = 1.0f / (1.0f + __expf(-v * scale));
          ((bf16_t*)C)[(size_t)gm * N + gn] = (bf16_t)v;
        } else {
          ((float*)C)[(size_t)gm * N + gn] = v;
        }
      }
}

// ---------------------------------------------------------------- host
extern "C" void kernel_launch(void* const* d_in, const int* in_sizes, int n_in,
                              void* d_out, int out_size, void* d_ws, size_t ws_size,
                              hipStream_t stream) {
  const float* q  = (const float*)d_in[0];
  const float* x  = (const float*)d_in[1];
  const float* Wq = (const float*)d_in[2];
  const float* bq = (const float*)d_in[3];
  const float* Wk = (const float*)d_in[4];
  const float* bk = (const float*)d_in[5];
  const float* Wv = (const float*)d_in[6];
  const float* bv = (const float*)d_in[7];

  bf16_t* qb  = (bf16_t*)d_ws;                       // [LQ,DIN]
  bf16_t* xb  = qb  + (size_t)LQ * DIN;              // [LKS,DIN]
  bf16_t* Wqt = xb  + (size_t)LKS * DIN;             // [DOUT,DIN] (W^T)
  bf16_t* Wkt = Wqt + (size_t)DIN * DOUT;
  bf16_t* Wvt = Wkt + (size_t)DIN * DOUT;
  bf16_t* Qb  = Wvt + (size_t)DIN * DOUT;            // [LQ,DOUT]
  bf16_t* Kb  = Qb  + (size_t)LQ * DOUT;             // [LKS,DOUT]
  bf16_t* Vb  = Kb  + (size_t)LKS * DOUT;            // [LKS,DOUT]
  bf16_t* Vt  = Vb  + (size_t)LKS * DOUT;            // [DOUT,LKS] (V^T)
  bf16_t* S   = Vt  + (size_t)LKS * DOUT;            // [LQ,LKS] 32MB, L2-resident

  dim3 blk(256);

  // f32 -> bf16 (q, x) ; fused transpose-convert for weights
  cvt_f32_to_bf16<<<(LQ * DIN) / 1024, blk, 0, stream>>>(q, qb, LQ * DIN);
  cvt_f32_to_bf16<<<(LKS * DIN) / 1024, blk, 0, stream>>>(x, xb, LKS * DIN);
  dim3 gW(DOUT / 32, DIN / 32);
  cvt_tr_f32_bf16<<<gW, blk, 0, stream>>>(Wq, Wqt, DIN, DOUT);
  cvt_tr_f32_bf16<<<gW, blk, 0, stream>>>(Wk, Wkt, DIN, DOUT);
  cvt_tr_f32_bf16<<<gW, blk, 0, stream>>>(Wv, Wvt, DIN, DOUT);

  // projections (TN: B = W^T), bias epilogue -> bf16
  dim3 gproj(DOUT / BN, LQ / BM);
  gemm_tn<0><<<gproj, blk, 0, stream>>>(qb, Wqt, bq, Qb, LQ, DOUT, DIN, 0.f);
  gemm_tn<0><<<gproj, blk, 0, stream>>>(xb, Wkt, bk, Kb, LKS, DOUT, DIN, 0.f);
  gemm_tn<0><<<gproj, blk, 0, stream>>>(xb, Wvt, bv, Vb, LKS, DOUT, DIN, 0.f);

  // V^T for the TN S@V GEMM
  dim3 gV(DOUT / 32, LKS / 32);
  tr_bf16<<<gV, blk, 0, stream>>>(Vb, Vt, LKS, DOUT);

  // S = sigmoid((Q @ K^T) / 32)   (K already [N,K] layout)
  dim3 gsc(LKS / BN, LQ / BM);
  gemm_tn<1><<<gsc, blk, 0, stream>>>(Qb, Kb, nullptr, S, LQ, LKS, DOUT, 0.03125f);

  // out = S @ V  ==  S[M,K] @ Vt[N,K]^T   -> f32
  dim3 gout(DOUT / BN, LQ / BM);
  gemm_tn<2><<<gout, blk, 0, stream>>>(S, Vt, nullptr, d_out, LQ, DOUT, LKS, 0.f);
}
